// SamplingMixtureDensityModule_1322849927197
// MI455X (gfx1250) — matrix operand order repaired and checked by reading, not compiled, not run on hardware
//
#include <hip/hip_runtime.h>
#include <hip/hip_bf16.h>
#include <stdint.h>

// ---------------------------------------------------------------------------
// Types for CDNA5 WMMA (wave32): v_wmma_f32_16x16x32_bf16
// ---------------------------------------------------------------------------
typedef __attribute__((ext_vector_type(16))) __bf16          v16bf;
typedef __attribute__((ext_vector_type(8)))  float           v8f;
typedef __attribute__((ext_vector_type(8)))  unsigned short  ushort8;

#define NROWS 100000
#define DDIM  512
#define MDIM  64

__device__ __forceinline__ unsigned short f32_bf16(float f) {
    unsigned int u = __float_as_uint(f);
    unsigned int r = u + 0x7FFFu + ((u >> 16) & 1u);   // round-to-nearest-even
    return (unsigned short)(r >> 16);
}
__device__ __forceinline__ float bf16_f32(unsigned short h) {
    return __uint_as_float(((unsigned int)h) << 16);
}

union BF16Frag { v16bf v; unsigned short u[16]; ushort8 h[2]; };

__device__ __forceinline__ v8f wmma_bf16(v16bf a, v16bf b, v8f c) {
    return __builtin_amdgcn_wmma_f32_16x16x32_bf16(
        /*neg_a=*/false, a, /*neg_b=*/false, b,
        /*c_mod=*/(short)0, c, /*reuse_a=*/false, /*reuse_b=*/false);
}

// 16 consecutive bf16 from 16B-aligned LDS/global memory -> B fragment.
__device__ __forceinline__ v16bf frag_load16(const unsigned short* p) {
    BF16Frag f;
    f.h[0] = *(const ushort8*)p;
    f.h[1] = *(const ushort8*)(p + 8);
    return f.v;
}

// A fragment (this file's contiguous k order, same as B): row pointer, khalf = lane>>4.
// elements 0..15 = K (khalf*16 .. +15)
__device__ __forceinline__ v16bf afrag_load(const unsigned short* row, int khalf) {
    BF16Frag f;
    f.h[0] = *(const ushort8*)(row + khalf * 16);
    f.h[1] = *(const ushort8*)(row + khalf * 16 + 8);
    return f.v;
}

// ---------------------------------------------------------------------------
// k0: pre-swizzle W [512][64] into WMMA B-fragment layout, bf16 hi + lo.
// Layout: frag[(c*4+mt)*32 + lane][e], e=0..15; c=K-chunk (16), mt=m-tile (4).
// B 32x16: lane&15 = column (m), K = (lane>>4)*16 + e.
// ---------------------------------------------------------------------------
__global__ void k0_prep_w(const float* __restrict__ W,
                          unsigned short* __restrict__ wfh,
                          unsigned short* __restrict__ wfl) {
    int t = blockIdx.x * blockDim.x + threadIdx.x;
    if (t >= 16 * 4 * 32) return;
    int lane  = t & 31;
    int mt    = (t >> 5) & 3;
    int c     = t >> 7;
    int m     = mt * 16 + (lane & 15);
    int khalf = lane >> 4;
    int base  = t * 16;
#pragma unroll
    for (int e = 0; e < 16; ++e) {
        int k = c * 32 + khalf * 16 + e;
        float w = W[k * MDIM + m];
        unsigned short h = f32_bf16(w);
        wfh[base + e] = h;
        wfl[base + e] = f32_bf16(w - bf16_f32(h));
    }
}

__global__ void kz_zero(float* __restrict__ p, int n) {
    int i = blockIdx.x * blockDim.x + threadIdx.x;
    if (i < n) p[i] = 0.0f;
}

// ---------------------------------------------------------------------------
// k1: gamma = softmax(x@W + b); also s[64] += col-sums; gammaT (bf16) to ws.
// 128 threads = 4 waves; each workgroup handles 64 rows of x.
// Wave w owns x-row tile [w*16, w*16+16) x all 4 m-tiles.
// ---------------------------------------------------------------------------
#define XSTR 40   // LDS row stride (bf16 elems): 80B = 20 banks, conflict-free
__global__ void __launch_bounds__(128, 1)
k1_gamma(const float* __restrict__ x, const float* __restrict__ b,
         const unsigned short* __restrict__ wfh,
         const unsigned short* __restrict__ wfl,
         unsigned short* __restrict__ gT,
         float* __restrict__ s_acc) {
    __shared__ unsigned short xh[64 * XSTR];
    __shared__ unsigned short xl[64 * XSTR];
    __shared__ unsigned short gtile[64 * 68];

    const int tid   = threadIdx.x;
    const int lane  = tid & 31;
    const int wave  = tid >> 5;       // 0..3
    const int n0    = blockIdx.x * 64;
    const int mlo   = lane & 15;
    const int rhalf = lane >> 4;

    v8f acc[4] = {};
    float bv[4];
#pragma unroll
    for (int mt = 0; mt < 4; ++mt) bv[mt] = b[mt * 16 + mlo];

    for (int c = 0; c < 16; ++c) {
        __syncthreads();
        // stage x[64 rows][32 cols of chunk c] as bf16 hi/lo
        {
            int row  = tid >> 1;
            int col0 = (tid & 1) * 16;
            int n    = n0 + row;
            const float* xp = x + (size_t)n * DDIM + c * 32 + col0;
            bool ok = (n < NROWS);
#pragma unroll
            for (int j = 0; j < 16; ++j) {
                float v = ok ? xp[j] : 0.0f;
                unsigned short vh = f32_bf16(v);
                xh[row * XSTR + col0 + j] = vh;
                xl[row * XSTR + col0 + j] = f32_bf16(v - bf16_f32(vh));
            }
        }
        __syncthreads();

        const unsigned short* arow_h = &xh[(wave * 16 + mlo) * XSTR];
        const unsigned short* arow_l = &xl[(wave * 16 + mlo) * XSTR];
        v16bf ah = afrag_load(arow_h, rhalf);
        v16bf al = afrag_load(arow_l, rhalf);
#pragma unroll
        for (int mt = 0; mt < 4; ++mt) {
            v16bf wh = *(const v16bf*)(wfh + ((c * 4 + mt) * 32 + lane) * 16);
            v16bf wl = *(const v16bf*)(wfl + ((c * 4 + mt) * 32 + lane) * 16);
            acc[mt] = wmma_bf16(ah, wh, acc[mt]);  // hi*hi
            acc[mt] = wmma_bf16(ah, wl, acc[mt]);  // hi*lo
            acc[mt] = wmma_bf16(al, wh, acc[mt]);  // lo*hi
        }
    }

    // row-wise softmax; row (within 16-row tile) = rhalf*8 + r, its 64 logits
    // live in the 16 lanes {rhalf*16 .. rhalf*16+15}, 4 per lane (one per mt).
    float gamma[4][8];
#pragma unroll
    for (int r = 0; r < 8; ++r) {
        float lmax = -3.4e38f;
#pragma unroll
        for (int mt = 0; mt < 4; ++mt) lmax = fmaxf(lmax, acc[mt][r] + bv[mt]);
#pragma unroll
        for (int m2 = 1; m2 < 16; m2 <<= 1) lmax = fmaxf(lmax, __shfl_xor(lmax, m2, 32));
        float rs = 0.0f;
#pragma unroll
        for (int mt = 0; mt < 4; ++mt) {
            float e = __expf(acc[mt][r] + bv[mt] - lmax);
            gamma[mt][r] = e;
            rs += e;
        }
#pragma unroll
        for (int m2 = 1; m2 < 16; m2 <<= 1) rs += __shfl_xor(rs, m2, 32);
        int nglob = n0 + wave * 16 + rhalf * 8 + r;
        float inv = (nglob < NROWS) ? (1.0f / rs) : 0.0f;  // mask padded rows
#pragma unroll
        for (int mt = 0; mt < 4; ++mt) gamma[mt][r] *= inv;
    }

    // s[m] accumulation (f32): lanes L and L+16 hold same m, different rows
#pragma unroll
    for (int mt = 0; mt < 4; ++mt) {
        float sl = 0.0f;
#pragma unroll
        for (int r = 0; r < 8; ++r) sl += gamma[mt][r];
        sl += __shfl_xor(sl, 16, 32);
        if (rhalf == 0) unsafeAtomicAdd(&s_acc[mt * 16 + mlo], sl);
    }

    // transpose gamma through LDS -> coalesced bf16 gammaT[64][NROWS] store
#pragma unroll
    for (int mt = 0; mt < 4; ++mt)
#pragma unroll
        for (int r = 0; r < 8; ++r)
            gtile[(mt * 16 + mlo) * 68 + wave * 16 + rhalf * 8 + r] =
                f32_bf16(gamma[mt][r]);
    __syncthreads();
    {
        int m = tid >> 1, half = tid & 1;
#pragma unroll
        for (int j = 0; j < 32; ++j) {
            int ncol = half * 32 + j;
            int ng = n0 + ncol;
            if (ng < NROWS)
                gT[(size_t)m * NROWS + ng] = gtile[m * 68 + ncol];
        }
    }
}

// ---------------------------------------------------------------------------
// k2: mu_acc = gammaT @ x ; ex2_acc = gammaT @ x*x  (bf16 hi/lo WMMA, f32 acc)
// 512 threads = 16 waves; persistent grid-stride over 3125 n-chunks of 32.
// Wave w: m-tile = w&3, d-group = w>>2 (4 d-tiles per 256-col half).
// ---------------------------------------------------------------------------
#define K2STR 40
#define K2_HALF_ELEMS (256 * K2STR)
__global__ void __launch_bounds__(512, 1)
k2_moments(const float* __restrict__ x,
           const unsigned short* __restrict__ gT,
           float* __restrict__ macc,
           float* __restrict__ e2acc) {
    extern __shared__ unsigned short lds[];
    unsigned short* xhT  = lds;
    unsigned short* xlT  = lds + 1 * K2_HALF_ELEMS;
    unsigned short* x2hT = lds + 2 * K2_HALF_ELEMS;
    unsigned short* x2lT = lds + 3 * K2_HALF_ELEMS;

    const int tid   = threadIdx.x;
    const int lane  = tid & 31;
    const int wave  = tid >> 5;     // 0..15
    const int mt    = wave & 3;
    const int dg    = wave >> 2;    // 0..3
    const int mlo   = lane & 15;
    const int khalf = lane >> 4;

    v8f accM[8] = {};
    v8f accE[8] = {};

    const int nChunks = NROWS / 32;  // 3125 exact
    for (int c = blockIdx.x; c < nChunks; c += gridDim.x) {
        // A fragment: gammaT[mt*16+mlo][c*32 + khalf*16 .. +15] (16B-aligned)
        BF16Frag af;
        {
            const unsigned short* gp =
                gT + (size_t)(mt * 16 + mlo) * NROWS + c * 32 + khalf * 16;
            af.h[0] = *(const ushort8*)gp;
            af.h[1] = *(const ushort8*)(gp + 8);
        }
#pragma unroll
        for (int half = 0; half < 2; ++half) {
            __syncthreads();
            // stage x[32 n][256 d] transposed: x, x^2 as bf16 hi/lo
            {
                int n  = tid >> 4;
                int d0 = (tid & 15) * 16;
                const float* xp =
                    x + (size_t)(c * 32 + n) * DDIM + half * 256 + d0;
#pragma unroll
                for (int j = 0; j < 16; ++j) {
                    float v  = xp[j];
                    float v2 = v * v;
                    unsigned short vh  = f32_bf16(v);
                    unsigned short v2h = f32_bf16(v2);
                    int idx = (d0 + j) * K2STR + n;
                    xhT[idx]  = vh;
                    xlT[idx]  = f32_bf16(v - bf16_f32(vh));
                    x2hT[idx] = v2h;
                    x2lT[idx] = f32_bf16(v2 - bf16_f32(v2h));
                }
            }
            __syncthreads();
#pragma unroll
            for (int dt = 0; dt < 4; ++dt) {
                int drow = (dg * 4 + dt) * 16 + mlo;         // local d column
                int bofs = drow * K2STR + khalf * 16;        // 16B aligned
                v16bf bxh = frag_load16(xhT  + bofs);
                v16bf bxl = frag_load16(xlT  + bofs);
                v16bf b2h = frag_load16(x2hT + bofs);
                v16bf b2l = frag_load16(x2lT + bofs);
                int ai = half * 4 + dt;
                accM[ai] = wmma_bf16(af.v, bxh, accM[ai]);
                accM[ai] = wmma_bf16(af.v, bxl, accM[ai]);
                accE[ai] = wmma_bf16(af.v, b2h, accE[ai]);
                accE[ai] = wmma_bf16(af.v, b2l, accE[ai]);
            }
        }
    }

    // flush partial sums (C/D layout: element r -> m-row khalf*8 + r)
#pragma unroll
    for (int half = 0; half < 2; ++half)
#pragma unroll
        for (int dt = 0; dt < 4; ++dt) {
            int d  = half * 256 + (dg * 4 + dt) * 16 + mlo;
            int ai = half * 4 + dt;
#pragma unroll
            for (int r = 0; r < 8; ++r) {
                int m = mt * 16 + khalf * 8 + r;
                unsafeAtomicAdd(&macc[m * DDIM + d],  accM[ai][r]);
                unsafeAtomicAdd(&e2acc[m * DDIM + d], accE[ai][r]);
            }
        }
}

// ---------------------------------------------------------------------------
// k3: finalize alpha, mu, sigma into d_out (alpha | mu | sigma, flat).
// ---------------------------------------------------------------------------
__global__ void k3_final(const float* __restrict__ s,
                         const float* __restrict__ macc,
                         const float* __restrict__ e2acc,
                         float* __restrict__ out) {
    int i = blockIdx.x * blockDim.x + threadIdx.x;
    if (i >= MDIM * DDIM) return;
    int m = i >> 9;
    float inv = 1.0f / s[m];
    float mu  = macc[i] * inv;
    float ex2 = e2acc[i] * inv;
    out[MDIM + i] = mu;
    out[MDIM + MDIM * DDIM + i] = ex2 - mu * mu;
    if (i < MDIM) out[i] = s[i] * (1.0f / (float)NROWS);
}

// ---------------------------------------------------------------------------
extern "C" void kernel_launch(void* const* d_in, const int* in_sizes, int n_in,
                              void* d_out, int out_size, void* d_ws, size_t ws_size,
                              hipStream_t stream) {
    const float* x = (const float*)d_in[0];
    const float* W = (const float*)d_in[1];
    const float* b = (const float*)d_in[2];
    float* out = (float*)d_out;

    // Workspace layout (bytes):
    //   [0, 65536)        W fragments hi (bf16)
    //   [65536, 131072)   W fragments lo (bf16)
    //   [131072, 131328)  s[64] f32
    //   [131328, 262400)  mu_acc [64][512] f32
    //   [262400, 393472)  ex2_acc [64][512] f32
    //   [393472, ...)     gammaT [64][100000] bf16 (12.8 MB)
    char* ws = (char*)d_ws;
    unsigned short* wfh = (unsigned short*)(ws);
    unsigned short* wfl = (unsigned short*)(ws + 65536);
    float* s_acc        = (float*)(ws + 131072);
    float* macc         = (float*)(ws + 131328);
    float* e2acc        = (float*)(ws + 262400);
    unsigned short* gT  = (unsigned short*)(ws + 393472);

    // zero s + mu_acc + ex2_acc (contiguous 65600 floats)
    kz_zero<<<(65600 + 255) / 256, 256, 0, stream>>>(s_acc, 65600);
    // pre-swizzle W into fragment layout
    k0_prep_w<<<8, 256, 0, stream>>>(W, wfh, wfl);
    // gamma + s + gammaT
    k1_gamma<<<(NROWS + 63) / 64, 128, 0, stream>>>(x, b, wfh, wfl, gT, s_acc);
    // weighted moments (dynamic LDS: 4 * 256 * 40 * 2 = 81920 B)
    k2_moments<<<256, 512, 4 * K2_HALF_ELEMS * (int)sizeof(unsigned short), stream>>>(
        x, gT, macc, e2acc);
    // finalize
    k3_final<<<(MDIM * DDIM + 255) / 256, 256, 0, stream>>>(s_acc, macc, e2acc, out);
}
